// MultiHeadSelfAttention_16595753632298
// MI455X (gfx1250) — matrix operand order repaired and checked by reading, not compiled, not run on hardware
//
#include <hip/hip_runtime.h>
#include <hip/hip_bf16.h>

#define BB 2
#define SS 2048
#define DD 1024
#define HH 16
#define HDD 64
#define MM (BB * SS) // 4096 tokens

typedef __bf16 bf16;
typedef __attribute__((ext_vector_type(16))) __bf16 v16bf;
typedef __attribute__((ext_vector_type(8)))  __bf16 v8bf;
typedef __attribute__((ext_vector_type(8)))  float  v8f;

// ---- WMMA fragment helpers (contiguous K order, same for A and B) ----

// A-matrix 16x32 bf16, row-major source with leading dim lda.
// lane row = lane&15 ; half-wave g = lane>>4 selects K subset:
// elems j -> K = g*16 + j  (one 32B load)
static __device__ __forceinline__ v16bf load_a_frag(const bf16* base, int lda, int lane) {
  const int row = lane & 15, g = lane >> 4;
  const bf16* p = base + (size_t)row * lda + g * 16;
  return *(const v16bf*)p;
}

// B-matrix 32x16 bf16 from "n-major" storage: row n holds contiguous K.
// lane col n = lane&15 ; elems j -> K = (lane>>4)*16 + j  (one 32B load)
static __device__ __forceinline__ v16bf load_b_frag(const bf16* base, int ldn, int lane) {
  const int n = lane & 15, g = lane >> 4;
  return *(const v16bf*)(base + (size_t)n * ldn + g * 16);
}

static __device__ __forceinline__ v8f wmma_bf16(v16bf a, v16bf b, v8f c) {
  return __builtin_amdgcn_wmma_f32_16x16x32_bf16(false, a, false, b, (short)0, c, false, false);
}

// ---- kernel 1: f32 -> bf16 conversion ----
__global__ void f32_to_bf16_kernel(const float* __restrict__ src, bf16* __restrict__ dst, int n) {
  int i = blockIdx.x * blockDim.x + threadIdx.x;
  if (i < n) dst[i] = (bf16)src[i];
}

// ---- kernel 2: fused QKV projection.  y = x @ W^T + b ----
// grid (2, 128, 3), block 256.  wave -> 32(m) x 64(n) tile: 8 WMMA / 6 loads per K step.
__global__ void qkv_proj_kernel(const bf16* __restrict__ xb, const bf16* __restrict__ wb,
                                const float* __restrict__ biasq, const float* __restrict__ biask,
                                const float* __restrict__ biasv,
                                bf16* __restrict__ qo, bf16* __restrict__ ko, bf16* __restrict__ vto) {
  const int lane = threadIdx.x & 31;
  const int wav  = threadIdx.x >> 5;
  const int g = lane >> 4, ln = lane & 15;
  const int which = blockIdx.z;
  const bf16* w = wb + (size_t)which * DD * DD;
  const float* bias = (which == 0) ? biasq : (which == 1) ? biask : biasv;
  const int m0 = blockIdx.y * 32;
  const int n0 = (blockIdx.x * 8 + wav) * 64;

  v8f c00 = {}, c01 = {}, c02 = {}, c03 = {};   // m-tile 0
  v8f c10 = {}, c11 = {}, c12 = {}, c13 = {};   // m-tile 1
  for (int k0 = 0; k0 < DD; k0 += 32) {
    v16bf a0 = load_a_frag(xb + (size_t)(m0     ) * DD + k0, DD, lane);
    v16bf a1 = load_a_frag(xb + (size_t)(m0 + 16) * DD + k0, DD, lane);
    v16bf b0 = load_b_frag(w + (size_t)(n0 +  0) * DD + k0, DD, lane);
    v16bf b1 = load_b_frag(w + (size_t)(n0 + 16) * DD + k0, DD, lane);
    v16bf b2 = load_b_frag(w + (size_t)(n0 + 32) * DD + k0, DD, lane);
    v16bf b3 = load_b_frag(w + (size_t)(n0 + 48) * DD + k0, DD, lane);
    c00 = wmma_bf16(a0, b0, c00);
    c01 = wmma_bf16(a0, b1, c01);
    c02 = wmma_bf16(a0, b2, c02);
    c03 = wmma_bf16(a0, b3, c03);
    c10 = wmma_bf16(a1, b0, c10);
    c11 = wmma_bf16(a1, b1, c11);
    c12 = wmma_bf16(a1, b2, c12);
    c13 = wmma_bf16(a1, b3, c13);
  }
#pragma unroll
  for (int t = 0; t < 2; ++t) {
#pragma unroll
    for (int r = 0; r < 8; ++r) {
      const int m = m0 + t * 16 + r + 8 * g;     // token row
      const int bidx = m / SS, s = m % SS;
#pragma unroll
      for (int j = 0; j < 4; ++j) {
        const int n = n0 + j * 16 + ln;
        float acc;
        if (t == 0) acc = (j == 0) ? c00[r] : (j == 1) ? c01[r] : (j == 2) ? c02[r] : c03[r];
        else        acc = (j == 0) ? c10[r] : (j == 1) ? c11[r] : (j == 2) ? c12[r] : c13[r];
        const float v = acc + bias[n];
        const int h = n >> 6, hd = n & 63;
        const bf16 vb = (bf16)v;
        if (which == 0)      qo [(((size_t)(bidx * HH + h)) * SS + s) * HDD + hd] = vb;
        else if (which == 1) ko [(((size_t)(bidx * HH + h)) * SS + s) * HDD + hd] = vb;
        else                 vto[(((size_t)(bidx * HH + h)) * HDD + hd) * SS + s] = vb; // V transposed
      }
    }
  }
}

// ---- kernel 3: causal flash attention.  grid (16, 16, 2), block 256 ----
__global__ void flash_attn_kernel(const bf16* __restrict__ qb, const bf16* __restrict__ kb,
                                  const bf16* __restrict__ vtb, bf16* __restrict__ ctx) {
  __shared__ bf16 pLds[8 * 16 * 32];       // per-wave private 16x32 P tile (8 KB)
  const int lane = threadIdx.x & 31;
  const int wav  = threadIdx.x >> 5;
  const int g = lane >> 4, ln = lane & 15;
  const int h = blockIdx.y, b = blockIdx.z;
  const int q0 = blockIdx.x * 128 + wav * 16;
  const bf16* Q  = qb  + ((size_t)(b * HH + h)) * SS * HDD;
  const bf16* K  = kb  + ((size_t)(b * HH + h)) * SS * HDD;
  const bf16* Vt = vtb + ((size_t)(b * HH + h)) * HDD * SS;
  bf16* pw = &pLds[wav * 512];

  const v16bf aq0 = load_a_frag(Q + (size_t)q0 * HDD +  0, HDD, lane);
  const v16bf aq1 = load_a_frag(Q + (size_t)q0 * HDD + 32, HDD, lane);

  v8f a0 = {}, a1 = {}, a2 = {}, a3 = {};
  float mrow[8], lrow[8];
#pragma unroll
  for (int r = 0; r < 8; ++r) { mrow[r] = -__builtin_inff(); lrow[r] = 0.f; }
  const float scale = 0.125f; // HD^-0.5 = 1/8

  for (int kv0 = 0; kv0 < q0 + 16; kv0 += 32) {
    v8f s0 = {}, s1 = {};
    {
      v16bf b00 = load_b_frag(K + (size_t)(kv0     ) * HDD +  0, HDD, lane);
      v16bf b01 = load_b_frag(K + (size_t)(kv0     ) * HDD + 32, HDD, lane);
      v16bf b10 = load_b_frag(K + (size_t)(kv0 + 16) * HDD +  0, HDD, lane);
      v16bf b11 = load_b_frag(K + (size_t)(kv0 + 16) * HDD + 32, HDD, lane);
      s0 = wmma_bf16(aq0, b00, s0); s0 = wmma_bf16(aq1, b01, s0);
      s1 = wmma_bf16(aq0, b10, s1); s1 = wmma_bf16(aq1, b11, s1);
    }
    const int n0c = kv0 + ln;
    const int n1c = n0c + 16;
#pragma unroll
    for (int r = 0; r < 8; ++r) {
      const int qrow = q0 + r + 8 * g;
      float v0 = (n0c <= qrow) ? s0[r] * scale : -__builtin_inff();
      float v1 = (n1c <= qrow) ? s1[r] * scale : -__builtin_inff();
      float t = fmaxf(v0, v1);
      t = fmaxf(t, __shfl_xor(t, 1, 32));
      t = fmaxf(t, __shfl_xor(t, 2, 32));
      t = fmaxf(t, __shfl_xor(t, 4, 32));
      t = fmaxf(t, __shfl_xor(t, 8, 32));     // row max across 16-lane group
      const float mn = fmaxf(mrow[r], t);
      const float alpha = __expf(mrow[r] - mn);
      mrow[r] = mn;
      const float p0 = __expf(v0 - mn);
      const float p1 = __expf(v1 - mn);
      float rs = p0 + p1;
      rs += __shfl_xor(rs, 1, 32);
      rs += __shfl_xor(rs, 2, 32);
      rs += __shfl_xor(rs, 4, 32);
      rs += __shfl_xor(rs, 8, 32);            // row sum
      lrow[r] = lrow[r] * alpha + rs;
      a0[r] *= alpha; a1[r] *= alpha; a2[r] *= alpha; a3[r] *= alpha;
      pw[(r + 8 * g) * 32 + ln     ] = (bf16)p0;   // D-layout -> row-major LDS
      pw[(r + 8 * g) * 32 + ln + 16] = (bf16)p1;
    }
    // same-wave LDS ops are in-order: no barrier needed for private region
    v16bf ap  = load_a_frag(pw, 32, lane);
    v16bf bv0 = load_b_frag(Vt + (size_t) 0 * SS + kv0, SS, lane);
    v16bf bv1 = load_b_frag(Vt + (size_t)16 * SS + kv0, SS, lane);
    v16bf bv2 = load_b_frag(Vt + (size_t)32 * SS + kv0, SS, lane);
    v16bf bv3 = load_b_frag(Vt + (size_t)48 * SS + kv0, SS, lane);
    a0 = wmma_bf16(ap, bv0, a0);
    a1 = wmma_bf16(ap, bv1, a1);
    a2 = wmma_bf16(ap, bv2, a2);
    a3 = wmma_bf16(ap, bv3, a3);
  }
#pragma unroll
  for (int r = 0; r < 8; ++r) {
    const float inv = 1.0f / lrow[r];
    const size_t tok  = (size_t)b * SS + q0 + r + 8 * g;
    const size_t base = tok * DD + (size_t)h * HDD;
    ctx[base +  0 + ln] = (bf16)(a0[r] * inv);
    ctx[base + 16 + ln] = (bf16)(a1[r] * inv);
    ctx[base + 32 + ln] = (bf16)(a2[r] * inv);
    ctx[base + 48 + ln] = (bf16)(a3[r] * inv);
  }
}

// ---- kernel 4: output projection, f32 result.  grid (2, 128) ----
__global__ void out_proj_kernel(const bf16* __restrict__ ctx, const bf16* __restrict__ wo,
                                const float* __restrict__ bo, float* __restrict__ out) {
  const int lane = threadIdx.x & 31;
  const int wav  = threadIdx.x >> 5;
  const int g = lane >> 4, ln = lane & 15;
  const int m0 = blockIdx.y * 32;
  const int n0 = (blockIdx.x * 8 + wav) * 64;

  v8f c00 = {}, c01 = {}, c02 = {}, c03 = {};
  v8f c10 = {}, c11 = {}, c12 = {}, c13 = {};
  for (int k0 = 0; k0 < DD; k0 += 32) {
    v16bf a0 = load_a_frag(ctx + (size_t)(m0     ) * DD + k0, DD, lane);
    v16bf a1 = load_a_frag(ctx + (size_t)(m0 + 16) * DD + k0, DD, lane);
    v16bf b0 = load_b_frag(wo + (size_t)(n0 +  0) * DD + k0, DD, lane);
    v16bf b1 = load_b_frag(wo + (size_t)(n0 + 16) * DD + k0, DD, lane);
    v16bf b2 = load_b_frag(wo + (size_t)(n0 + 32) * DD + k0, DD, lane);
    v16bf b3 = load_b_frag(wo + (size_t)(n0 + 48) * DD + k0, DD, lane);
    c00 = wmma_bf16(a0, b0, c00);
    c01 = wmma_bf16(a0, b1, c01);
    c02 = wmma_bf16(a0, b2, c02);
    c03 = wmma_bf16(a0, b3, c03);
    c10 = wmma_bf16(a1, b0, c10);
    c11 = wmma_bf16(a1, b1, c11);
    c12 = wmma_bf16(a1, b2, c12);
    c13 = wmma_bf16(a1, b3, c13);
  }
#pragma unroll
  for (int t = 0; t < 2; ++t) {
#pragma unroll
    for (int r = 0; r < 8; ++r) {
      const size_t m = m0 + t * 16 + r + 8 * g;
#pragma unroll
      for (int j = 0; j < 4; ++j) {
        const int n = n0 + j * 16 + ln;
        float acc;
        if (t == 0) acc = (j == 0) ? c00[r] : (j == 1) ? c01[r] : (j == 2) ? c02[r] : c03[r];
        else        acc = (j == 0) ? c10[r] : (j == 1) ? c11[r] : (j == 2) ? c12[r] : c13[r];
        out[m * DD + n] = acc + bo[n];
      }
    }
  }
}

extern "C" void kernel_launch(void* const* d_in, const int* in_sizes, int n_in,
                              void* d_out, int out_size, void* d_ws, size_t ws_size,
                              hipStream_t stream) {
  const float* x  = (const float*)d_in[0];
  const float* wq = (const float*)d_in[1];
  const float* bq = (const float*)d_in[2];
  const float* wk = (const float*)d_in[3];
  const float* bk = (const float*)d_in[4];
  const float* wv = (const float*)d_in[5];
  const float* bv = (const float*)d_in[6];
  const float* wo = (const float*)d_in[7];
  const float* bo = (const float*)d_in[8];
  float* out = (float*)d_out;

  // workspace layout (bf16), total 48 MB
  char* ws = (char*)d_ws;
  bf16* xb  = (bf16*)(ws);                       //  8 MB : x bf16 [4096,1024]
  bf16* wqb = (bf16*)(ws + (size_t) 8  * 1024 * 1024);  // wq,wk,wv,wo bf16 (8 MB)
  bf16* wkb = wqb + (size_t)DD * DD;
  bf16* wvb = wkb + (size_t)DD * DD;
  bf16* wob = wvb + (size_t)DD * DD;
  bf16* qb  = (bf16*)(ws + (size_t)16 * 1024 * 1024);   // Q  [B,H,S,HD]
  bf16* kb2 = (bf16*)(ws + (size_t)24 * 1024 * 1024);   // K  [B,H,S,HD]
  bf16* vtb = (bf16*)(ws + (size_t)32 * 1024 * 1024);   // Vt [B,H,HD,S]
  bf16* ctx = (bf16*)(ws + (size_t)40 * 1024 * 1024);   // ctx [4096,1024]

  const int nx = MM * DD;        // 4194304
  const int nw = DD * DD;        // 1048576
  f32_to_bf16_kernel<<<dim3((nx + 255) / 256), dim3(256), 0, stream>>>(x,  xb,  nx);
  f32_to_bf16_kernel<<<dim3((nw + 255) / 256), dim3(256), 0, stream>>>(wq, wqb, nw);
  f32_to_bf16_kernel<<<dim3((nw + 255) / 256), dim3(256), 0, stream>>>(wk, wkb, nw);
  f32_to_bf16_kernel<<<dim3((nw + 255) / 256), dim3(256), 0, stream>>>(wv, wvb, nw);
  f32_to_bf16_kernel<<<dim3((nw + 255) / 256), dim3(256), 0, stream>>>(wo, wob, nw);

  qkv_proj_kernel<<<dim3(2, MM / 32, 3), dim3(256), 0, stream>>>(xb, wqb, bq, bk, bv, qb, kb2, vtb);
  flash_attn_kernel<<<dim3(SS / 128, HH, BB), dim3(256), 0, stream>>>(qb, kb2, vtb, ctx);
  out_proj_kernel<<<dim3(2, MM / 32, 1), dim3(256), 0, stream>>>(ctx, wob, bo, out);
}
